// BertSelfAttention_61065845014529
// MI455X (gfx1250) — matrix operand order repaired and checked by reading, not compiled, not run on hardware
//
#include <hip/hip_runtime.h>

typedef __attribute__((ext_vector_type(16))) __bf16 v16bf;
typedef __attribute__((ext_vector_type(8)))  float  v8f;

union Frag {
    uint4        q[2];
    unsigned int u[8];
    v16bf        v;
};

__device__ __forceinline__ unsigned short f2bf(float f) {
    unsigned int u = __builtin_bit_cast(unsigned int, f);
    u += 0x7FFFu + ((u >> 16) & 1u);          // round-to-nearest-even
    return (unsigned short)(u >> 16);
}
__device__ __forceinline__ float bflo(unsigned int u) { return __builtin_bit_cast(float, u << 16); }
__device__ __forceinline__ float bfhi(unsigned int u) { return __builtin_bit_cast(float, u & 0xFFFF0000u); }

// ---------------------------------------------------------------------------
// Batched bf16-WMMA GEMM:  C[z] = A[z] (f32, row-major, converted to bf16)
//                               @ B[z] (bf16, PRE-TRANSPOSED: BT[n][k])  (+bias)
// Block tile 128(M) x 64(N), K-step 64. 8 waves: waveM in [0,4), waveN in [0,2),
// each wave owns a 32x32 patch = 2x2 v_wmma_f32_16x16x32_bf16 accumulators.
// Requires M%128==0, N%64==0, K%64==0 (true for every call below).
// C batch offset = (z/cDiv)*cSO + (z%cDiv)*cSI   (handles [B,S,H*Dh] scatter).
// ---------------------------------------------------------------------------
__launch_bounds__(256)
__global__ void gemm_bf16(const float* __restrict__ A,
                          const unsigned short* __restrict__ BT,
                          void* __restrict__ Cv,
                          const float* __restrict__ bias,
                          int M, int N, int K,
                          int lda, int ldb, int ldc,
                          long aStride, long bStride,
                          int cDiv, long cSO, long cSI,
                          int outBf16)
{
    __shared__ unsigned int lds_a[128][36];   // 128 rows x 32 dwords (64 bf16), padded
    __shared__ unsigned int lds_b[64][36];    // 64 rows  x 32 dwords, padded

    const int tid   = threadIdx.x;
    const int lane  = tid & 31;
    const int wave  = tid >> 5;
    const int waveM = wave & 3;
    const int waveN = wave >> 2;
    const int kh    = lane >> 4;              // K-half select (ISA A/B layouts)
    const int lr    = lane & 15;

    const int  z   = blockIdx.z;
    const long m0  = (long)blockIdx.x * 128;
    const long n0  = (long)blockIdx.y * 64;
    const float*        Ab = A + (long)z * aStride;
    const unsigned int* Bb = (const unsigned int*)BT + (((long)z * bStride) >> 1);
    const long cOff = (long)(z / cDiv) * cSO + (long)(z % cDiv) * cSI;

    v8f acc[2][2] = {};

    const int ktiles = K >> 6;
    for (int kt = 0; kt < ktiles; ++kt) {
        // --- stage A tile (convert f32 -> packed bf16 dwords) ---
        #pragma unroll
        for (int i = 0; i < 16; ++i) {
            int idx = tid + i * 256;              // 4096 dwords
            int r = idx >> 5, c = idx & 31;
            const float* src = Ab + (m0 + r) * lda + (kt << 6) + (c << 1);
            if (i == 0 && kt + 1 < ktiles)
                __builtin_prefetch(src + 64, 0, 1);   // global_prefetch next K-tile
            float f0 = src[0], f1 = src[1];
            lds_a[r][c] = (unsigned int)f2bf(f0) | ((unsigned int)f2bf(f1) << 16);
        }
        // --- stage B tile (already bf16) ---
        #pragma unroll
        for (int i = 0; i < 8; ++i) {
            int idx = tid + i * 256;              // 2048 dwords
            int r = idx >> 5, c = idx & 31;
            lds_b[r][c] = Bb[(((long)(n0 + r) * ldb + (kt << 6)) >> 1) + c];
        }
        __syncthreads();

        #pragma unroll
        for (int kk = 0; kk < 2; ++kk) {          // two K=32 WMMA steps per tile
            const int kd = kk << 4;               // dword offset within staged row
            Frag fa[2], fb[2];
            #pragma unroll
            for (int t = 0; t < 2; ++t) {
                // A frag: lane lr holds row; dwords [kh*4 .. +3] and [8+kh*4 .. +3]
                const unsigned int* ar = &lds_a[waveM * 32 + t * 16 + lr][kd];
                fa[t].q[0] = *(const uint4*)(ar + kh * 4);
                fa[t].q[1] = *(const uint4*)(ar + 8 + kh * 4);
                // B frag: lane lr holds column n; dwords [kh*4 .. +3] and [8+kh*4 .. +3]
                const unsigned int* br = &lds_b[waveN * 32 + t * 16 + lr][kd];
                fb[t].q[0] = *(const uint4*)(br + kh * 4);
                fb[t].q[1] = *(const uint4*)(br + 8 + kh * 4);
            }
            #pragma unroll
            for (int tm = 0; tm < 2; ++tm)
                #pragma unroll
                for (int tn = 0; tn < 2; ++tn)
                    acc[tm][tn] = __builtin_amdgcn_wmma_f32_16x16x32_bf16(
                        false, fa[tm].v, false, fb[tn].v,
                        (short)0, acc[tm][tn], false, false);
        }
        __syncthreads();
    }

    // --- epilogue: C layout M = j + 8*kh, N = lr ---
    #pragma unroll
    for (int tm = 0; tm < 2; ++tm) {
        #pragma unroll
        for (int tn = 0; tn < 2; ++tn) {
            long col = n0 + waveN * 32 + tn * 16 + lr;
            float badd = bias ? bias[col] : 0.0f;
            #pragma unroll
            for (int j = 0; j < 8; ++j) {
                long row  = m0 + waveM * 32 + tm * 16 + kh * 8 + j;
                float val = acc[tm][tn][j] + badd;
                long cidx = cOff + row * ldc + col;
                if (outBf16) ((unsigned short*)Cv)[cidx] = f2bf(val);
                else         ((float*)Cv)[cidx]          = val;
            }
        }
    }
}

// ---------------------------------------------------------------------------
// Transpose + convert weight [K x N] f32 -> WT [N x K] bf16
// ---------------------------------------------------------------------------
__global__ void transpose_w(const float* __restrict__ W, unsigned short* __restrict__ WT,
                            int K, int N)
{
    long idx = (long)blockIdx.x * 256 + threadIdx.x;
    if (idx < (long)K * N) {
        long k = idx / N;
        long n = idx - k * N;
        WT[n * (long)K + k] = f2bf(W[idx]);
    }
}

__global__ void concat_bias(const float* __restrict__ bq, const float* __restrict__ bk,
                            const float* __restrict__ bv, const float* __restrict__ bpv,
                            float* __restrict__ out)
{
    int i = blockIdx.x * 256 + threadIdx.x;
    if (i < 2368) {
        float v;
        if (i < 768)       v = bq[i];
        else if (i < 1536) v = bk[i - 768];
        else if (i < 2304) v = bv[i - 1536];
        else               v = bpv[i - 2304];
        out[i] = v;
    }
}

// Repack v and pv (from fused qkv f32 [256 x 2368]) into transposed bf16 for
// use as WMMA B operands: vT[b,h,d,ks], pvT[b,d,ks]
__global__ void repack_vpv(const float* __restrict__ qkv,
                           unsigned short* __restrict__ vT,
                           unsigned short* __restrict__ pvT)
{
    int idx = blockIdx.x * 256 + threadIdx.x;
    if (idx < 196608) {                       // vT: B*H*64*128
        int ks = idx & 127;
        int tt = idx >> 7;
        int d  = tt & 63;
        int bh = tt >> 6;
        int b  = bh / 12, h = bh - b * 12;
        vT[idx] = f2bf(qkv[(long)(b * 128 + ks) * 2368 + 1536 + h * 64 + d]);
    } else if (idx < 196608 + 16384) {        // pvT: B*64*128
        int i2 = idx - 196608;
        int ks = i2 & 127;
        int tt = i2 >> 7;
        int d  = tt & 63;
        int b  = tt >> 6;
        pvT[i2] = f2bf(qkv[(long)(b * 128 + ks) * 2368 + 2304 + d]);
    }
}

// ---------------------------------------------------------------------------
// Scores with torch-.view semantics: within batch b, flat (h,q,k,d) == flat
// (sq,sk,c), so ip row R (1536 bf16 channels) yields the 12 score elements
// e = R*12 + t  with ra = ip[R, 64t:64t+64], rb = ip[R, 768+64t:...],
// h = e>>14, q = (e>>7)&127, k = e&127.
// One block stages 16 ip rows (48 KB LDS); 192 threads compute 16x12 scores.
// ---------------------------------------------------------------------------
__launch_bounds__(256)
__global__ void scores_kernel(const unsigned short* __restrict__ ip_chunk, // [8192][1536] bf16
                              const float* __restrict__ qkv,               // [256][2368] f32
                              const float* __restrict__ mask,              // [B,1,1,S]
                              float* __restrict__ scores,                  // [B,H,S,S]
                              int chunk)
{
    __shared__ unsigned int lds[16 * 768];    // 16 rows x 1536 bf16 = 48 KB
    const int tid = threadIdx.x;
    const unsigned int* src = (const unsigned int*)ip_chunk;
    const int lrow0 = blockIdx.x * 16;
    #pragma unroll
    for (int r = 0; r < 16; ++r)
        #pragma unroll
        for (int i = 0; i < 3; ++i)
            lds[r * 768 + i * 256 + tid] = src[(long)(lrow0 + r) * 768 + i * 256 + tid];
    __syncthreads();

    if (tid < 192) {
        const int lr    = tid / 12;
        const int t     = tid - lr * 12;
        const int row_g = chunk * 8192 + lrow0 + lr;      // global ip row (b,sq,sk)
        const int b     = row_g >> 14;                    // 16384 rows per batch
        const int e     = (row_g & 16383) * 12 + t;
        const int h     = e >> 14;
        const int qq    = (e >> 7) & 127;
        const int kk    = e & 127;
        const unsigned int* ra = &lds[lr * 768 + t * 32];
        const unsigned int* rb = ra + 384;
        const float* qp = qkv + (long)(b * 128 + qq) * 2368 + h * 64;
        const float* kp = qkv + (long)(b * 128 + kk) * 2368 + 768 + h * 64;
        float acc = 0.f;
        #pragma unroll
        for (int j = 0; j < 32; ++j) {
            unsigned int da = ra[j], db = rb[j];
            float qe0 = qp[2 * j]     + bflo(da);
            float qe1 = qp[2 * j + 1] + bfhi(da);
            float ke0 = kp[2 * j]     + bflo(db);
            float ke1 = kp[2 * j + 1] + bfhi(db);
            acc += qe0 * ke0 + qe1 * ke1;
        }
        scores[(((long)(b * 12 + h) * 128 + qq) << 7) + kk] = acc * 0.125f + mask[b * 128 + kk];
    }
}

// one wave32 per row of 128; shfl_xor tree reductions
__launch_bounds__(256)
__global__ void softmax_kernel(const float* __restrict__ scores, float* __restrict__ probs)
{
    const int lane = threadIdx.x & 31;
    const long row = (long)blockIdx.x * 8 + (threadIdx.x >> 5);
    const float* s = scores + row * 128;
    float v[4];
    #pragma unroll
    for (int i = 0; i < 4; ++i) v[i] = s[lane + 32 * i];
    float m = fmaxf(fmaxf(v[0], v[1]), fmaxf(v[2], v[3]));
    #pragma unroll
    for (int off = 16; off > 0; off >>= 1) m = fmaxf(m, __shfl_xor(m, off, 32));
    float e[4], sum = 0.f;
    #pragma unroll
    for (int i = 0; i < 4; ++i) { e[i] = __expf(v[i] - m); sum += e[i]; }
    #pragma unroll
    for (int off = 16; off > 0; off >>= 1) sum += __shfl_xor(sum, off, 32);
    float inv = 1.0f / sum;
    float* p = probs + row * 128;
    #pragma unroll
    for (int i = 0; i < 4; ++i) p[lane + 32 * i] = e[i] * inv;
}

// ---------------------------------------------------------------------------
extern "C" void kernel_launch(void* const* d_in, const int* in_sizes, int n_in,
                              void* d_out, int out_size, void* d_ws, size_t ws_size,
                              hipStream_t stream)
{
    (void)in_sizes; (void)n_in; (void)out_size; (void)ws_size;
    const float* hs    = (const float*)d_in[0];
    const float* amask = (const float*)d_in[1];
    const float* ipath = (const float*)d_in[2];
    const float* span  = (const float*)d_in[3];
    const float* Wq  = (const float*)d_in[4];   const float* bq  = (const float*)d_in[5];
    const float* Wk  = (const float*)d_in[6];   const float* bk  = (const float*)d_in[7];
    const float* Wv  = (const float*)d_in[8];   const float* bvp = (const float*)d_in[9];
    const float* Wpv = (const float*)d_in[10];  const float* bpv = (const float*)d_in[11];
    const float* Wip = (const float*)d_in[12];
    const float* Wmlp= (const float*)d_in[13];  const float* bmlp= (const float*)d_in[14];
    float* out = (float*)d_out;

    char* ws = (char*)d_ws;
    size_t o = 0;
    auto alloc = [&](size_t bytes) -> char* {
        char* p = ws + o;
        o += (bytes + 255) & ~(size_t)255;
        return p;
    };
    unsigned short* WT_qkvp = (unsigned short*)alloc(2368ull * 768 * 2);
    unsigned short* WT_ip   = (unsigned short*)alloc(1536ull * 768 * 2);
    unsigned short* WT_mlp  = (unsigned short*)alloc(768ull * 832 * 2);
    float*          bias_c  = (float*)alloc(2368ull * 4);
    float*          qkv     = (float*)alloc(256ull * 2368 * 4);
    unsigned short* vT      = (unsigned short*)alloc(196608ull * 2);
    unsigned short* pvT     = (unsigned short*)alloc(16384ull * 2);
    unsigned short* ip_buf  = (unsigned short*)alloc(8192ull * 1536 * 2); // 25 MB, L2-resident
    float*          scores  = (float*)alloc(393216ull * 4);
    float*          probs   = (float*)alloc(393216ull * 4);
    float*          ctx     = (float*)alloc(2ull * 128 * 832 * 4);

    dim3 blk(256);
    // weights -> transposed bf16
    transpose_w<<<(768 * 768 + 255) / 256, blk, 0, stream>>>(Wq,  WT_qkvp,                768, 768);
    transpose_w<<<(768 * 768 + 255) / 256, blk, 0, stream>>>(Wk,  WT_qkvp + 768 * 768,    768, 768);
    transpose_w<<<(768 * 768 + 255) / 256, blk, 0, stream>>>(Wv,  WT_qkvp + 2 * 768 * 768,768, 768);
    transpose_w<<<(768 * 64  + 255) / 256, blk, 0, stream>>>(Wpv, WT_qkvp + 3 * 768 * 768,768, 64);
    transpose_w<<<(768 * 1536 + 255) / 256, blk, 0, stream>>>(Wip,  WT_ip,  768, 1536);
    transpose_w<<<(832 * 768  + 255) / 256, blk, 0, stream>>>(Wmlp, WT_mlp, 832, 768);
    concat_bias<<<(2368 + 255) / 256, blk, 0, stream>>>(bq, bk, bvp, bpv, bias_c);

    // fused q|k|v|pv projection: [256x768] @ [768x2368] (+bias) -> f32
    gemm_bf16<<<dim3(2, 37, 1), blk, 0, stream>>>(hs, WT_qkvp, qkv, bias_c,
        256, 2368, 768, 768, 768, 2368, 0, 0, 1, 0, 0, 0);
    repack_vpv<<<832, blk, 0, stream>>>(qkv, vT, pvT);

    // dominant GEMM (77 GF), chunked so the bf16 intermediate stays in L2
    for (int c = 0; c < 4; ++c) {
        gemm_bf16<<<dim3(64, 24, 1), blk, 0, stream>>>(ipath + (long)c * 8192 * 768,
            WT_ip, ip_buf, nullptr,
            8192, 1536, 768, 768, 768, 1536, 0, 0, 1, 0, 0, 1);
        scores_kernel<<<512, blk, 0, stream>>>(ip_buf, qkv, amask, scores, c);
    }

    softmax_kernel<<<384, blk, 0, stream>>>(scores, probs);

    // ctx[b,h] = probs[b,h] @ v[b,h], scattered into concat layout [B,S,832]
    gemm_bf16<<<dim3(1, 1, 24), blk, 0, stream>>>(probs, vT, ctx, nullptr,
        128, 64, 128, 128, 128, 832, 16384, 8192, 12, 106496, 64, 0);
    // parse_ctx[b] = span[b] @ pv[b] -> columns 768..831
    gemm_bf16<<<dim3(1, 1, 2), blk, 0, stream>>>(span, pvT, ctx + 768, nullptr,
        128, 64, 128, 128, 128, 832, 16384, 8192, 1, 106496, 0, 0);
    // out = ctx_cat [256x832] @ Wmlp + bmlp
    gemm_bf16<<<dim3(2, 12, 1), blk, 0, stream>>>(ctx, WT_mlp, out, bmlp,
        256, 768, 832, 832, 832, 768, 0, 0, 1, 0, 0, 0);
}
